// SparseSelfAttention_74096775791007
// MI455X (gfx1250) — matrix operand order repaired and checked by reading, not compiled, not run on hardware
//
#include <hip/hip_runtime.h>
#include <hip/hip_bf16.h>

// ---------------------------------------------------------------------------
// Types for CDNA5 WMMA
// ---------------------------------------------------------------------------
typedef __attribute__((ext_vector_type(16))) __bf16 v16bf;
typedef __attribute__((ext_vector_type(8)))  float  v8f;
typedef __attribute__((ext_vector_type(8)))  unsigned short us8;
typedef unsigned short ushort_t;
typedef int v4i_gl __attribute__((__vector_size__(16)));   // GCC-style int4 (matches builtin sig)

#define AS1 __attribute__((address_space(1)))
#define AS3 __attribute__((address_space(3)))

struct BF16x16 { us8 lo, hi; };   // 32 bytes, reinterpreted as v16bf

__device__ __forceinline__ unsigned short f2bf(float f) {
  unsigned u = __builtin_bit_cast(unsigned, f);
  u += 0x7FFFu + ((u >> 16) & 1u);      // round-to-nearest-even
  return (unsigned short)(u >> 16);
}

// ---------------------------------------------------------------------------
// CDNA5 async global->LDS copy path (guarded; falls back to direct stores)
// ---------------------------------------------------------------------------
#if defined(__gfx1250__) && __has_builtin(__builtin_amdgcn_global_load_async_to_lds_b128)
#define USE_ASYNC 1
#else
#define USE_ASYNC 0
#endif

__device__ __forceinline__ void async_cp16(const ushort_t* g, ushort_t* l) {
#if USE_ASYNC
  __builtin_amdgcn_global_load_async_to_lds_b128(
      (AS1 v4i_gl*)g, (AS3 v4i_gl*)l, 0, 0);
#else
  *(us8*)l = *(const us8*)g;
#endif
}

__device__ __forceinline__ void wait_async_le4() {
#if USE_ASYNC
#if __has_builtin(__builtin_amdgcn_s_wait_asynccnt)
  __builtin_amdgcn_s_wait_asynccnt((short)4);
#else
  asm volatile("s_wait_asynccnt 0x4" ::: "memory");
#endif
#endif
}
__device__ __forceinline__ void wait_async_0() {
#if USE_ASYNC
#if __has_builtin(__builtin_amdgcn_s_wait_asynccnt)
  __builtin_amdgcn_s_wait_asynccnt((short)0);
#else
  asm volatile("s_wait_asynccnt 0x0" ::: "memory");
#endif
#endif
}

// ---------------------------------------------------------------------------
// Tiled bf16 WMMA GEMM:  C[M,N] = A[M,K] @ Bt[N,K]^T   (A, Bt already bf16)
// BM=128, BN=128, BK=32, 256 threads = 8 waves (grid 4x2),
// each wave computes 32x64 = 2x4 WMMA 16x16 tiles. Double-buffered LDS with
// GLOBAL_LOAD_ASYNC_TO_LDS_B128 staging (ASYNCcnt) when available.
// EPI: 0 = none, 1 = +bias,ReLU, 2 = +bias.  OUTBF: 1 = write bf16 C.
// ---------------------------------------------------------------------------
#define BM 128
#define BN 128
#define BKK 32
#define SA 40   // padded LDS stride in bf16 units (80B, 16B-aligned chunks)

template <int EPI, int OUTBF>
__global__ __launch_bounds__(256) void wmma_gemm_kernel(
    const ushort_t* __restrict__ A,   // [M,K] bf16 row-major
    const ushort_t* __restrict__ Bt,  // [N,K] bf16 row-major (B transposed)
    const float* __restrict__ bias,
    float* __restrict__ Cf, ushort_t* __restrict__ Cb,
    int M, int N, int K) {
  __shared__ __align__(16) ushort_t lA[2][BM * SA];
  __shared__ __align__(16) ushort_t lB[2][BN * SA];

  const int tid    = threadIdx.x;
  const int lane   = tid & 31;
  const int wave   = tid >> 5;
  const int waveM  = wave >> 1;       // 0..3  (32 rows each)
  const int waveN  = wave & 1;        // 0..1  (64 cols each)
  const int lane16 = lane & 15;
  const int hi     = lane >> 4;       // 0 or 1
  const int m0     = blockIdx.y * BM;
  const int n0     = blockIdx.x * BN;

  v8f acc[2][4] = {};

  // ---- stage one K-block into buffer `buf` (4 async b128 per wave) ----
  auto stage = [&](int buf, int kb) {
#pragma unroll
    for (int i = 0; i < 2; ++i) {
      int c  = tid + i * 256;         // 0..511
      int r  = c >> 2;                // 0..127
      int kc = (c & 3) << 3;          // 0,8,16,24
      int gr = m0 + r; if (gr >= M) gr = M - 1;   // clamp: keep per-wave async count uniform
      async_cp16(A  + (size_t)gr       * K + kb + kc, &lA[buf][r * SA + kc]);
      async_cp16(Bt + (size_t)(n0 + r) * K + kb + kc, &lB[buf][r * SA + kc]);
    }
  };

  const int nk = K / BKK;
  stage(0, 0);

  for (int it = 0; it < nk; ++it) {
    const int cur = it & 1;
    if (it + 1 < nk) { stage(cur ^ 1, (it + 1) * BKK); wait_async_le4(); }
    else             { wait_async_0(); }
    __syncthreads();                  // all waves' copies for `cur` complete

    const ushort_t* la = &lA[cur][0];
    const ushort_t* lb = &lB[cur][0];

    BF16x16 au[2], bu[4];
#pragma unroll
    for (int mi = 0; mi < 2; ++mi) {
      int row = waveM * 32 + mi * 16 + lane16;
      int c0  = hi * 8;               // lanes 0-15: K0..7/K16..23 ; 16-31: K8..15/K24..31
      au[mi].lo = *(const us8*)&la[row * SA + c0];
      au[mi].hi = *(const us8*)&la[row * SA + c0 + 16];
    }
#pragma unroll
    for (int ni = 0; ni < 4; ++ni) {
      int col = waveN * 64 + ni * 16 + lane16;
      int k0  = hi * 8;               // lanes 0-15: K0..7/K16..23 ; 16-31: K8..15/K24..31
      bu[ni].lo = *(const us8*)&lb[col * SA + k0];
      bu[ni].hi = *(const us8*)&lb[col * SA + k0 + 16];
    }
#pragma unroll
    for (int mi = 0; mi < 2; ++mi) {
      v16bf af = __builtin_bit_cast(v16bf, au[mi]);
#pragma unroll
      for (int ni = 0; ni < 4; ++ni) {
        v16bf bf = __builtin_bit_cast(v16bf, bu[ni]);
        acc[mi][ni] = __builtin_amdgcn_wmma_f32_16x16x32_bf16(
            false, af, false, bf, (short)0, acc[mi][ni], false, false);
      }
    }
    __syncthreads();                  // `cur` buffer free for restage
  }

  // ---- epilogue: C/D layout: VGPR i -> row (hi*8 + i), col = lane16 ----
#pragma unroll
  for (int mi = 0; mi < 2; ++mi) {
#pragma unroll
    for (int ni = 0; ni < 4; ++ni) {
      int col = n0 + waveN * 64 + ni * 16 + lane16;
      float bv = 0.f;
      if (EPI > 0) bv = bias[col];
#pragma unroll
      for (int i = 0; i < 8; ++i) {
        int row = m0 + waveM * 32 + mi * 16 + hi * 8 + i;
        if (row < M) {
          float val = acc[mi][ni][i] + bv;
          if (EPI == 1) val = fmaxf(val, 0.f);
          if (OUTBF) Cb[(size_t)row * N + col] = f2bf(val);
          else       Cf[(size_t)row * N + col] = val;
        }
      }
    }
  }
}

// ---------------------------------------------------------------------------
// Precision-conversion / transpose helpers (one-shot, tiny vs GEMM traffic)
// ---------------------------------------------------------------------------
__global__ __launch_bounds__(256) void cvt_bf16_kernel(
    const float* __restrict__ in, ushort_t* __restrict__ out, long long n4) {
  long long i = (long long)blockIdx.x * 256 + threadIdx.x;
  if (i < n4) {
    float4 v = ((const float4*)in)[i];
    uint2 p;
    p.x = (unsigned)f2bf(v.x) | ((unsigned)f2bf(v.y) << 16);
    p.y = (unsigned)f2bf(v.z) | ((unsigned)f2bf(v.w) << 16);
    ((uint2*)out)[i] = p;
  }
}

// Wt[n][k] = bf16(W[k][n]); K,N multiples of 32
__global__ __launch_bounds__(256) void transpose_bf16_kernel(
    const float* __restrict__ W, ushort_t* __restrict__ Wt, int K, int N) {
  __shared__ float tile[32][33];
  int tx = threadIdx.x & 31, ty = threadIdx.x >> 5;   // 32x8
  int n = blockIdx.x * 32 + tx;
#pragma unroll
  for (int i = 0; i < 32; i += 8) {
    int k = blockIdx.y * 32 + ty + i;
    tile[ty + i][tx] = W[(size_t)k * N + n];
  }
  __syncthreads();
  int k2 = blockIdx.y * 32 + tx;
#pragma unroll
  for (int i = 0; i < 32; i += 8) {
    int n2 = blockIdx.x * 32 + ty + i;
    Wt[(size_t)n2 * K + k2] = f2bf(tile[tx][ty + i]);
  }
}

// ---------------------------------------------------------------------------
// Elementwise / edge kernels
// ---------------------------------------------------------------------------
__global__ __launch_bounds__(256) void fill_kernel(float* p, float v, long long n) {
  long long i = (long long)blockIdx.x * 256 + threadIdx.x;
  if (i < n) p[i] = v;
}

__device__ __forceinline__ void atomicMaxF(float* addr, float val) {
  unsigned* ua = (unsigned*)addr;
  unsigned old = *ua;
  while (__builtin_bit_cast(float, old) < val) {
    unsigned assumed = old;
    old = atomicCAS(ua, assumed, __builtin_bit_cast(unsigned, val));
    if (old == assumed) break;
  }
}

__global__ __launch_bounds__(256) void edge_score_kernel(
    const float* __restrict__ q, const float* __restrict__ k,
    const int* __restrict__ src, const int* __restrict__ dst,
    float* __restrict__ e, int E_) {
  int t = blockIdx.x * 256 + threadIdx.x;
  if (t >= E_ * 8) return;
  int edge = t >> 3, head = t & 7;
  int s = src[edge], d = dst[edge];
  const float4* kp = (const float4*)(k + (size_t)s * 512 + head * 64);
  const float4* qp = (const float4*)(q + (size_t)d * 512 + head * 64);
  float sum = 0.f;
#pragma unroll
  for (int i = 0; i < 16; ++i) {
    float4 a = kp[i], b = qp[i];
    sum += a.x * b.x + a.y * b.y + a.z * b.z + a.w * b.w;
  }
  e[t] = sum * 0.125f;    // 1/sqrt(64)
}

__global__ __launch_bounds__(256) void edge_max_kernel(
    const float* __restrict__ e, const int* __restrict__ dst,
    float* __restrict__ m, int E_) {
  int t = blockIdx.x * 256 + threadIdx.x;
  if (t >= E_ * 8) return;
  int edge = t >> 3, head = t & 7;
  atomicMaxF(&m[dst[edge] * 8 + head], e[t]);
}

__global__ __launch_bounds__(256) void edge_expsum_kernel(
    float* __restrict__ e, const int* __restrict__ dst,
    const float* __restrict__ m, float* __restrict__ z, int E_) {
  int t = blockIdx.x * 256 + threadIdx.x;
  if (t >= E_ * 8) return;
  int edge = t >> 3, head = t & 7;
  int d = dst[edge];
  float ex = __expf(e[t] - m[d * 8 + head]);
  e[t] = ex;
  atomicAdd(&z[d * 8 + head], ex);
}

__global__ __launch_bounds__(256) void edge_scatter_kernel(
    const float* __restrict__ ex, const float* __restrict__ z,
    const float* __restrict__ v, const int* __restrict__ src,
    const int* __restrict__ dst, float* __restrict__ a, int E_) {
  int t = blockIdx.x * 256 + threadIdx.x;
  if (t >= E_ * 8) return;
  int edge = t >> 3, head = t & 7;
  int s = src[edge], d = dst[edge];
  float attn = ex[t] / z[d * 8 + head];
  const float4* vp = (const float4*)(v + (size_t)s * 512 + head * 64);
  float* ap = a + (size_t)d * 512 + head * 64;
#pragma unroll
  for (int i = 0; i < 16; ++i) {
    float4 vv = vp[i];
    atomicAdd(ap + i * 4 + 0, attn * vv.x);
    atomicAdd(ap + i * 4 + 1, attn * vv.y);
    atomicAdd(ap + i * 4 + 2, attn * vv.z);
    atomicAdd(ap + i * 4 + 3, attn * vv.w);
  }
}

// out = LayerNorm(A + Bd) * g + b; optional bf16 shadow copy in outb
__global__ __launch_bounds__(256) void add_ln_kernel(
    const float* __restrict__ A, const float* __restrict__ Bd,
    const float* __restrict__ g, const float* __restrict__ bb,
    float* __restrict__ out, ushort_t* __restrict__ outb, int Mrows) {
  int row = blockIdx.x;
  if (row >= Mrows) return;
  __shared__ float s1[256], s2[256];
  const float* pa = A + (size_t)row * 512;
  const float* pb = Bd + (size_t)row * 512;
  int t = threadIdx.x;
  float v0 = pa[t] + pb[t];
  float v1 = pa[t + 256] + pb[t + 256];
  s1[t] = v0 + v1;
  s2[t] = v0 * v0 + v1 * v1;
  __syncthreads();
  for (int s = 128; s > 0; s >>= 1) {
    if (t < s) { s1[t] += s1[t + s]; s2[t] += s2[t + s]; }
    __syncthreads();
  }
  float mu  = s1[0] * (1.f / 512.f);
  float var = s2[0] * (1.f / 512.f) - mu * mu;
  float rs  = rsqrtf(var + 1e-5f);
  float r0 = (v0 - mu) * rs * g[t] + bb[t];
  float r1 = (v1 - mu) * rs * g[t + 256] + bb[t + 256];
  out[(size_t)row * 512 + t]       = r0;
  out[(size_t)row * 512 + t + 256] = r1;
  if (outb) {
    outb[(size_t)row * 512 + t]       = f2bf(r0);
    outb[(size_t)row * 512 + t + 256] = f2bf(r1);
  }
}

// ---------------------------------------------------------------------------
// Launcher
// ---------------------------------------------------------------------------
static inline int cdiv_i(int a, int b) { return (a + b - 1) / b; }

extern "C" void kernel_launch(void* const* d_in, const int* in_sizes, int n_in,
                              void* d_out, int out_size, void* d_ws, size_t ws_size,
                              hipStream_t stream) {
  const float* h    = (const float*)d_in[0];
  const int*   src  = (const int*)d_in[1];
  const int*   dst  = (const int*)d_in[2];
  const float* Wq   = (const float*)d_in[3];
  const float* Wk   = (const float*)d_in[4];
  const float* Wv   = (const float*)d_in[5];
  const float* Wo   = (const float*)d_in[6];
  const float* ln1g = (const float*)d_in[7];
  const float* ln1b = (const float*)d_in[8];
  const float* ln2g = (const float*)d_in[9];
  const float* ln2b = (const float*)d_in[10];
  const float* W1   = (const float*)d_in[11];
  const float* b1   = (const float*)d_in[12];
  const float* W2   = (const float*)d_in[13];
  const float* b2   = (const float*)d_in[14];

  const int DM  = 512, Hh = 8, DFF = in_sizes[12];   // 2048
  const int N_  = in_sizes[0] / DM;                  // 20000
  const int E_  = in_sizes[1];                       // 640000
  const size_t nDM = (size_t)N_ * DM;

  // bump allocator over workspace (256B aligned blocks)
  char* wp = (char*)d_ws;
  auto alloc = [&](size_t bytes) -> void* {
    void* r = wp; wp += (bytes + 255) & ~(size_t)255; return r;
  };
  float*    q   = (float*)alloc(nDM * 4);
  float*    kk  = (float*)alloc(nDM * 4);
  float*    vv  = (float*)alloc(nDM * 4);
  float*    e   = (float*)alloc((size_t)E_ * Hh * 4);
  float*    m   = (float*)alloc((size_t)N_ * Hh * 4);
  float*    z   = (float*)alloc((size_t)N_ * Hh * 4);
  float*    a   = (float*)alloc(nDM * 4);
  float*    o   = (float*)alloc(nDM * 4);           // reused as f2
  float*    x   = (float*)alloc(nDM * 4);
  ushort_t* hb  = (ushort_t*)alloc(nDM * 2);
  ushort_t* ab  = (ushort_t*)alloc(nDM * 2);
  ushort_t* xb  = (ushort_t*)alloc(nDM * 2);
  ushort_t* f1b = (ushort_t*)alloc((size_t)N_ * DFF * 2);
  ushort_t* Wqt = (ushort_t*)alloc((size_t)DM * DM * 2);
  ushort_t* Wkt = (ushort_t*)alloc((size_t)DM * DM * 2);
  ushort_t* Wvt = (ushort_t*)alloc((size_t)DM * DM * 2);
  ushort_t* Wot = (ushort_t*)alloc((size_t)DM * DM * 2);
  ushort_t* W1t = (ushort_t*)alloc((size_t)DM * DFF * 2);
  ushort_t* W2t = (ushort_t*)alloc((size_t)DFF * DM * 2);
  float*    f2  = o;
  float*    out = (float*)d_out;

  dim3 blk(256);

  // ---- one-shot precision prep ----
  cvt_bf16_kernel<<<(int)((nDM / 4 + 255) / 256), blk, 0, stream>>>(h, hb, (long long)(nDM / 4));
  transpose_bf16_kernel<<<dim3(DM / 32, DM / 32), blk, 0, stream>>>(Wq, Wqt, DM, DM);
  transpose_bf16_kernel<<<dim3(DM / 32, DM / 32), blk, 0, stream>>>(Wk, Wkt, DM, DM);
  transpose_bf16_kernel<<<dim3(DM / 32, DM / 32), blk, 0, stream>>>(Wv, Wvt, DM, DM);
  transpose_bf16_kernel<<<dim3(DM / 32, DM / 32), blk, 0, stream>>>(Wo, Wot, DM, DM);
  transpose_bf16_kernel<<<dim3(DFF / 32, DM / 32), blk, 0, stream>>>(W1, W1t, DM, DFF);
  transpose_bf16_kernel<<<dim3(DM / 32, DFF / 32), blk, 0, stream>>>(W2, W2t, DFF, DM);

  dim3 gN512(DM / BN,  cdiv_i(N_, BM));
  dim3 gNDFF(DFF / BN, cdiv_i(N_, BM));

  // ---- q/k/v projections ----
  wmma_gemm_kernel<0, 0><<<gN512, blk, 0, stream>>>(hb, Wqt, nullptr, q,  nullptr, N_, DM, DM);
  wmma_gemm_kernel<0, 0><<<gN512, blk, 0, stream>>>(hb, Wkt, nullptr, kk, nullptr, N_, DM, DM);
  wmma_gemm_kernel<0, 0><<<gN512, blk, 0, stream>>>(hb, Wvt, nullptr, vv, nullptr, N_, DM, DM);

  // ---- init segment buffers ----
  long long nMH = (long long)N_ * Hh;
  fill_kernel<<<cdiv_i((int)nMH, 256), blk, 0, stream>>>(m, -1e30f, nMH);
  fill_kernel<<<cdiv_i((int)nMH, 256), blk, 0, stream>>>(z, 0.f, nMH);
  fill_kernel<<<(int)((nDM + 255) / 256), blk, 0, stream>>>(a, 0.f, (long long)nDM);

  // ---- sparse attention (edge phase) ----
  int nEH = E_ * Hh;
  edge_score_kernel  <<<cdiv_i(nEH, 256), blk, 0, stream>>>(q, kk, src, dst, e, E_);
  edge_max_kernel    <<<cdiv_i(nEH, 256), blk, 0, stream>>>(e, dst, m, E_);
  edge_expsum_kernel <<<cdiv_i(nEH, 256), blk, 0, stream>>>(e, dst, m, z, E_);
  edge_scatter_kernel<<<cdiv_i(nEH, 256), blk, 0, stream>>>(e, z, vv, src, dst, a, E_);

  // ---- output projection + residual LN1 (x also emitted as bf16) ----
  cvt_bf16_kernel<<<(int)((nDM / 4 + 255) / 256), blk, 0, stream>>>(a, ab, (long long)(nDM / 4));
  wmma_gemm_kernel<0, 0><<<gN512, blk, 0, stream>>>(ab, Wot, nullptr, o, nullptr, N_, DM, DM);
  add_ln_kernel<<<N_, blk, 0, stream>>>(h, o, ln1g, ln1b, x, xb, N_);

  // ---- FFN (f1 kept bf16-only) + residual LN2 ----
  wmma_gemm_kernel<1, 1><<<gNDFF, blk, 0, stream>>>(xb, W1t, b1, nullptr, f1b, N_, DFF, DM);
  wmma_gemm_kernel<2, 0><<<gN512, blk, 0, stream>>>(f1b, W2t, b2, f2, nullptr, N_, DM, DFF);
  add_ln_kernel<<<N_, blk, 0, stream>>>(x, f2, ln2g, ln2b, out, nullptr, N_);
}
